// _VGG16Head_17910013624620
// MI455X (gfx1250) — matrix operand order repaired and checked by reading, not compiled, not run on hardware
//
#include <hip/hip_runtime.h>
#include <hip/hip_bf16.h>
#include <math.h>

// ---------------- types for WMMA ----------------
typedef __attribute__((ext_vector_type(16))) __bf16 v16bf;
typedef __attribute__((ext_vector_type(8)))  float  v8f;

union FragBF {            // 16 bf16 = 32 bytes, loaded as two 16B chunks from LDS
    uint4 u[2];
    v16bf v;
};

// ---------------- small helpers ----------------
__device__ __forceinline__ unsigned short f2bf(float f) {
    unsigned int u = __float_as_uint(f);
    u += 0x7FFFu + ((u >> 16) & 1u);          // round-to-nearest-even
    return (unsigned short)(u >> 16);
}
__device__ __forceinline__ float bf2f(unsigned short h) {
    return __uint_as_float(((unsigned int)h) << 16);
}
// low 32 bits of a flat shared-space address == LDS byte address (ISA aperture rule)
__device__ __forceinline__ unsigned lds_off(const void* p) {
    return (unsigned)(uintptr_t)p;
}
// CDNA5 async copy: global -> LDS, 16B per lane, tracked by ASYNCcnt
__device__ __forceinline__ void async_ld_b128(unsigned lds_byte_addr, const void* g) {
    asm volatile("global_load_async_to_lds_b128 %0, %1, off"
                 :: "v"(lds_byte_addr), "v"(g) : "memory");
}

// ---------------- problem constants ----------------
#define RR    1024
#define CC    512
#define HH    50
#define WW    50
#define PP    7
#define DD    25088      // C*P*P
#define HID   4096
#define NCLS  21
#define NDELT 84         // NCLS*4

// =====================================================================
// 1) ROI max pool -> bf16 activation matrix X [R, D]
// =====================================================================
__global__ void roi_pool_bf16(const float* __restrict__ fmap,
                              const float* __restrict__ rois,
                              const int*   __restrict__ img_size,
                              unsigned short* __restrict__ X) {
    int idx = blockIdx.x * blockDim.x + threadIdx.x;
    const int total = RR * CC * PP * PP;
    if (idx >= total) return;
    int pw = idx % PP;
    int t  = idx / PP;
    int ph = t % PP;
    t /= PP;
    int c  = t % CC;
    int r  = t / CC;

    float scale = (float)HH / (float)img_size[0];          // 50/800
    float x1 = roundf(rois[r * 4 + 0] * scale);
    float y1 = roundf(rois[r * 4 + 1] * scale);
    float x2 = roundf(rois[r * 4 + 2] * scale);
    float y2 = roundf(rois[r * 4 + 3] * scale);
    float bh = fmaxf(y2 - y1 + 1.0f, 1.0f) * (1.0f / (float)PP);
    float bw = fmaxf(x2 - x1 + 1.0f, 1.0f) * (1.0f / (float)PP);

    int hs = (int)fminf(fmaxf(floorf((float)ph * bh) + y1, 0.0f), (float)HH);
    int he = (int)fminf(fmaxf(ceilf((float)(ph + 1) * bh) + y1, 0.0f), (float)HH);
    int ws = (int)fminf(fmaxf(floorf((float)pw * bw) + x1, 0.0f), (float)WW);
    int we = (int)fminf(fmaxf(ceilf((float)(pw + 1) * bw) + x1, 0.0f), (float)WW);

    float m = -INFINITY;
    const float* fc = fmap + (size_t)c * (HH * WW);
    for (int h = hs; h < he; ++h)
        for (int w = ws; w < we; ++w)
            m = fmaxf(m, fc[h * WW + w]);
    if (he <= hs || we <= ws) m = 0.0f;                    // empty bin -> 0

    X[(size_t)r * DD + c * (PP * PP) + ph * PP + pw] = f2bf(m);
}

// =====================================================================
// 2) transpose + downconvert:  W[K,N] fp32  ->  Wt[N,K] bf16
// =====================================================================
__global__ void transpose_f32_bf16(const float* __restrict__ W,
                                   unsigned short* __restrict__ Wt,
                                   int K, int N) {
    __shared__ float tile[32][33];
    int n0 = blockIdx.x * 32;
    int k0 = blockIdx.y * 32;
    int tx = threadIdx.x, ty = threadIdx.y;                // (32, 8)
    for (int i = ty; i < 32; i += 8) {
        int k = k0 + i, n = n0 + tx;
        tile[i][tx] = (k < K && n < N) ? W[(size_t)k * N + n] : 0.0f;
    }
    __syncthreads();
    for (int i = ty; i < 32; i += 8) {
        int n = n0 + i, k = k0 + tx;
        if (n < N && k < K)
            Wt[(size_t)n * K + k] = f2bf(tile[tx][i]);
    }
}

// =====================================================================
// 3) GEMM:  Out[M,N] = relu(A[M,K] * Bt[N,K]^T + bias)   (bf16 in/out)
//    256 threads (8 waves, 4x2), tile 128x128x32, double-buffered LDS
//    filled with GLOBAL_LOAD_ASYNC_TO_LDS_B128 (ASYNCcnt pipeline).
// =====================================================================
#define BM 128
#define BN 128
#define BK 32
#define LDP 40   // padded row stride in halfs (80B) to dodge bank conflicts

__global__ void __launch_bounds__(256)
gemm_bf16_wmma(const unsigned short* __restrict__ A,
               const unsigned short* __restrict__ Bt,
               const float* __restrict__ bias,
               unsigned short* __restrict__ Out,
               int M, int N, int K, int do_relu) {
    __shared__ __align__(16) unsigned short As[2][BM * LDP];
    __shared__ __align__(16) unsigned short Bs[2][BN * LDP];

    const int tid  = threadIdx.x;
    const int lane = tid & 31;
    const int wave = tid >> 5;
    const int wm   = wave >> 1;          // 0..3  -> 32-row strip
    const int wn   = wave & 1;           // 0..1  -> 64-col strip
    const int mlan = lane & 15;
    const int sel  = lane >> 4;          // 0/1 half-wave select

    const int bn = blockIdx.x * BN;
    const int bm = blockIdx.y * BM;

    v8f acc[2][4];
#pragma unroll
    for (int i = 0; i < 2; ++i)
#pragma unroll
        for (int j = 0; j < 4; ++j)
            acc[i][j] = (v8f){};

    const int lrow = tid >> 2;           // 0..63
    const int lcol = (tid & 3) * 8;      // 0,8,16,24 halfs

    // issue one tile's async copies (4 ASYNCcnt increments per wave)
    auto issue_tile = [&](int buf, int k0) {
#pragma unroll
        for (int i = 0; i < 2; ++i) {
            int r = lrow + i * 64;
            async_ld_b128(lds_off(&As[buf][r * LDP + lcol]),
                          &A[(size_t)(bm + r) * K + k0 + lcol]);
            async_ld_b128(lds_off(&Bs[buf][r * LDP + lcol]),
                          &Bt[(size_t)(bn + r) * K + k0 + lcol]);
        }
    };

    issue_tile(0, 0);
    int buf = 0;
    for (int k0 = 0; k0 < K; k0 += BK) {
        if (k0 + BK < K) {
            issue_tile(buf ^ 1, k0 + BK);                 // prefetch next tile
            asm volatile("s_wait_asynccnt 0x4" ::: "memory");  // current tile done
        } else {
            asm volatile("s_wait_asynccnt 0x0" ::: "memory");
        }
        __syncthreads();                                   // all waves' data visible

        // ---- fragments per documented 16-bit WMMA VGPR layouts ----
        FragBF a[2], b[4];
#pragma unroll
        for (int i = 0; i < 2; ++i) {
            const unsigned short* p = &As[buf][(wm * 32 + i * 16 + mlan) * LDP];
            a[i].u[0] = *(const uint4*)(p + sel * 8);        // K = sel*8 .. +8
            a[i].u[1] = *(const uint4*)(p + 16 + sel * 8);   // K = 16+sel*8 ..
        }
#pragma unroll
        for (int j = 0; j < 4; ++j) {
            const unsigned short* p = &Bs[buf][(wn * 64 + j * 16 + mlan) * LDP];
            b[j].u[0] = *(const uint4*)(p + sel * 8);        // K = sel*8 .. +8
            b[j].u[1] = *(const uint4*)(p + 16 + sel * 8);   // K = 16+sel*8 ..
        }
#pragma unroll
        for (int i = 0; i < 2; ++i)
#pragma unroll
            for (int j = 0; j < 4; ++j)
                acc[i][j] = __builtin_amdgcn_wmma_f32_16x16x32_bf16(
                    false, a[i].v, false, b[j].v,
                    (short)0, acc[i][j], false, false);

        __syncthreads();   // all waves done reading buf -> safe to overwrite next iter
        buf ^= 1;
    }

    // ---- epilogue: bias + relu, f32 -> bf16, per documented C/D layout ----
#pragma unroll
    for (int i = 0; i < 2; ++i) {
        int tm = bm + wm * 32 + i * 16;
#pragma unroll
        for (int j = 0; j < 4; ++j) {
            int tn = bn + wn * 64 + j * 16 + mlan;
            float bv = bias[tn];
#pragma unroll
            for (int rr = 0; rr < 8; ++rr) {
                float v = acc[i][j][rr] + bv;
                if (do_relu) v = fmaxf(v, 0.0f);
                Out[(size_t)(tm + rr + 8 * sel) * N + tn] = f2bf(v);
            }
        }
    }
}

// =====================================================================
// 4) heads + softmax: one block per ROI (128 thr), mid2 row in LDS
// =====================================================================
__global__ void __launch_bounds__(128)
head_softmax(const unsigned short* __restrict__ mid2,
             const float* __restrict__ Wd, const float* __restrict__ bd,
             const float* __restrict__ Wp, const float* __restrict__ bp,
             float* __restrict__ out_delta, float* __restrict__ out_prob) {
    __shared__ __align__(16) unsigned short row[HID];
    __shared__ float sm[NCLS];
    const int r = blockIdx.x;
    const int j = threadIdx.x;

    const uint4* src = (const uint4*)(mid2 + (size_t)r * HID);
    for (int i = j; i < HID / 8; i += 128)
        ((uint4*)row)[i] = src[i];
    __syncthreads();

    if (j < NDELT) {
        float acc = 0.0f;
        for (int k = 0; k < HID; ++k)
            acc += bf2f(row[k]) * Wd[(size_t)k * NDELT + j];
        out_delta[(size_t)r * NDELT + j] = acc + bd[j];
    } else if (j < NDELT + NCLS) {
        int jj = j - NDELT;
        float acc = 0.0f;
        for (int k = 0; k < HID; ++k)
            acc += bf2f(row[k]) * Wp[(size_t)k * NCLS + jj];
        sm[jj] = acc + bp[jj];
    }
    __syncthreads();
    if (j >= NDELT && j < NDELT + NCLS) {
        int jj = j - NDELT;
        float mx = -INFINITY;
        for (int q = 0; q < NCLS; ++q) mx = fmaxf(mx, sm[q]);
        float s = 0.0f;
        for (int q = 0; q < NCLS; ++q) s += expf(sm[q] - mx);
        out_prob[(size_t)r * NCLS + jj] = expf(sm[jj] - mx) / s;
    }
}

// =====================================================================
// launch
// =====================================================================
extern "C" void kernel_launch(void* const* d_in, const int* in_sizes, int n_in,
                              void* d_out, int out_size, void* d_ws, size_t ws_size,
                              hipStream_t stream) {
    const float* fmap = (const float*)d_in[0];
    const float* rois = (const float*)d_in[1];
    const float* W1   = (const float*)d_in[2];
    const float* b1   = (const float*)d_in[3];
    const float* W2   = (const float*)d_in[4];
    const float* b2   = (const float*)d_in[5];
    const float* Wd   = (const float*)d_in[6];
    const float* bd   = (const float*)d_in[7];
    const float* Wp   = (const float*)d_in[8];
    const float* bp   = (const float*)d_in[9];
    const int*   img  = (const int*)d_in[10];
    float* out        = (float*)d_out;

    unsigned short* ws = (unsigned short*)d_ws;
    unsigned short* Xbf  = ws;                                   // [R, D]
    unsigned short* W1t  = Xbf  + (size_t)RR * DD;               // [HID, D]
    unsigned short* W2t  = W1t  + (size_t)HID * DD;              // [HID, HID]
    unsigned short* mid1 = W2t  + (size_t)HID * HID;             // [R, HID]
    unsigned short* mid2 = mid1 + (size_t)RR * HID;              // [R, HID]

    // 1) ROI pool -> bf16 X
    {
        int total = RR * CC * PP * PP;
        roi_pool_bf16<<<(total + 255) / 256, 256, 0, stream>>>(fmap, rois, img, Xbf);
    }
    // 2) weight transpose + downconvert (once per launch)
    transpose_f32_bf16<<<dim3(HID / 32, DD / 32),  dim3(32, 8), 0, stream>>>(W1, W1t, DD,  HID);
    transpose_f32_bf16<<<dim3(HID / 32, HID / 32), dim3(32, 8), 0, stream>>>(W2, W2t, HID, HID);
    // 3) fc1: [1024,25088] @ [25088,4096] + b1, relu
    gemm_bf16_wmma<<<dim3(HID / BN, RR / BM), 256, 0, stream>>>(
        Xbf, W1t, b1, mid1, RR, HID, DD, 1);
    // 4) fc2: [1024,4096] @ [4096,4096] + b2, relu
    gemm_bf16_wmma<<<dim3(HID / BN, RR / BM), 256, 0, stream>>>(
        mid1, W2t, b2, mid2, RR, HID, HID, 1);
    // 5) heads + softmax -> concatenated (delta[1024*84], prob[1024*21])
    head_softmax<<<RR, 128, 0, stream>>>(
        mid2, Wd, bd, Wp, bp, out, out + (size_t)RR * NDELT);
}
